// TSMEncoderLayer_50946902065874
// MI455X (gfx1250) — matrix operand order repaired and checked by reading, not compiled, not run on hardware
//
#include <hip/hip_runtime.h>
#include <hip/hip_bf16.h>
#include <stdint.h>

#define BATCH   4
#define SEQ     2048
#define DMODEL  256
#define DINNER  512
#define DSTATE  16
#define DCONV   4
#define DTRANK  16
#define ROWS    (BATCH*SEQ)          /* 8192 */
#define XCOLS   (DTRANK + 2*DSTATE)  /* 48 */
#define EPSV    1e-5f

typedef __attribute__((ext_vector_type(16))) __bf16 v16bf;
typedef __attribute__((ext_vector_type(8)))  __bf16 v8bf;
typedef __attribute__((ext_vector_type(8)))  float  v8f;

// ---------------- conversion kernels ----------------
__global__ void cvt_bf16_kernel(const float* __restrict__ in, __bf16* __restrict__ out, int n) {
  int i = blockIdx.x * blockDim.x + threadIdx.x;
  if (i < n) out[i] = (__bf16)in[i];
}

// W is [N,K] row-major fp32 -> Wt [K,N] bf16 (so B-fragment loads are contiguous)
__global__ void cvt_tr_bf16_kernel(const float* __restrict__ in, __bf16* __restrict__ out, int N, int K) {
  int i = blockIdx.x * blockDim.x + threadIdx.x;
  if (i < N * K) {
    int n = i / K, k = i % K;
    out[(size_t)k * N + n] = (__bf16)in[i];
  }
}

// =====================================================================
// Block-cooperative WMMA GEMM with async-LDS double buffering.
// Block = 256 threads (8 waves). Tile: 128(M) x 64(N); K template-unrolled.
// B tile (32x64 bf16, rows padded to 72 elems) staged via
// global_load_async_to_lds_b128 (ASYNCcnt), double buffered.
// =====================================================================
#define BROW 72   /* padded LDS row length in elements (144B, 16B-aligned rows) */

template<int K, bool RESID>
__global__ void __launch_bounds__(256)
gemm_lds_kernel(const __bf16* __restrict__ A, const __bf16* __restrict__ Bt,
                float* __restrict__ C, const float* __restrict__ resid, int M, int N) {
  __shared__ __align__(128) __bf16 sh[2][32][BROW];
  constexpr int NK = K / 32;

  const int tid  = threadIdx.x;
  const int w    = tid >> 5;         // wave 0..7
  const int lane = tid & 31;
  const int h    = lane >> 4;
  const int nlo  = lane & 15;

  const int nG   = N / 64;
  const int mblk = blockIdx.x / nG;
  const int ng   = blockIdx.x % nG;
  const int colBase = ng * 64;
  const int mrow0   = mblk * 128 + w * 16;

  // per-thread async staging coordinates: 16 bytes each, 32 rows x 8 segments
  const int skk  = tid >> 3;   // 0..31 : k within tile
  const int sseg = tid & 7;    // 0..7  : 8-element segment within 64-col row
  const uint32_t ldsRowOff = (uint32_t)(skk * (BROW * 2) + sseg * 16);
  const uint32_t ldsBuf0   = (uint32_t)(uintptr_t)&sh[0][0][0];
  const uint32_t ldsBuf1   = (uint32_t)(uintptr_t)&sh[1][0][0];
  const __bf16*  gstage    = Bt + (size_t)skk * N + colBase + sseg * 8;

  v8f acc[4];
  #pragma unroll
  for (int t = 0; t < 4; t++)
    #pragma unroll
    for (int i = 0; i < 8; i++) acc[t][i] = 0.f;

  const __bf16* aRow = A + (size_t)(mrow0 + nlo) * K;

  // ---- prologue: stage k0=0 into buf0, preload A regs for k0=0 ----
  asm volatile("global_load_async_to_lds_b128 %0, %1, off"
               :: "v"(ldsBuf0 + ldsRowOff),
                  "v"((uint64_t)(uintptr_t)gstage) : "memory");
  v8bf alo = *(const v8bf*)(aRow + 0 + 8 * h);
  v8bf ahi = *(const v8bf*)(aRow + 16 + 8 * h);

  #pragma unroll
  for (int i = 0; i < NK; i++) {
    const int k0 = i * 32;
    constexpr bool LAST_CHECK = true; (void)LAST_CHECK;
    const bool hasNext = (i + 1 < NK);

    // issue async stage for next K-step into the other buffer
    if (hasNext) {
      const __bf16* gnext = gstage + (size_t)(k0 + 32) * N;
      uint32_t dst = ((i + 1) & 1) ? ldsBuf1 : ldsBuf0;
      asm volatile("global_load_async_to_lds_b128 %0, %1, off"
                   :: "v"(dst + ldsRowOff),
                      "v"((uint64_t)(uintptr_t)gnext) : "memory");
      asm volatile("s_wait_asynccnt 0x1" ::: "memory");   // stage i complete
    } else {
      asm volatile("s_wait_asynccnt 0x0" ::: "memory");
    }
    __syncthreads();                                      // LDS tile visible to all waves

    // prefetch A for next K-step (register double buffer)
    v8bf anlo, anhi;
    if (hasNext) {
      anlo = *(const v8bf*)(aRow + k0 + 32 + 8 * h);
      anhi = *(const v8bf*)(aRow + k0 + 48 + 8 * h);
    }

    v16bf afrag;
    #pragma unroll
    for (int e = 0; e < 8; e++) { afrag[e] = alo[e]; afrag[8 + e] = ahi[e]; }

    // B fragments from LDS: column n = lane%16, k rows in table order (8h+e, 16+8h+e)
    const __bf16 (*sb)[BROW] = sh[i & 1];
    #pragma unroll
    for (int t = 0; t < 4; t++) {
      v16bf bfrag;
      #pragma unroll
      for (int e = 0; e < 8; e++) { bfrag[e] = sb[8 * h + e][t * 16 + nlo]; bfrag[8 + e] = sb[16 + 8 * h + e][t * 16 + nlo]; }
      acc[t] = __builtin_amdgcn_wmma_f32_16x16x32_bf16(
          false, afrag, false, bfrag, (short)0, acc[t], false, false);
    }
    __syncthreads();      // all waves done reading this buffer before it is re-staged
    alo = anlo; ahi = anhi;
  }

  #pragma unroll
  for (int t = 0; t < 4; t++) {
    int n = colBase + t * 16 + nlo;
    #pragma unroll
    for (int r = 0; r < 8; r++) {
      int m = mrow0 + 8 * h + r;
      size_t idx = (size_t)m * N + n;
      float v = acc[t][r];
      if (RESID) v += resid[idx];
      C[idx] = v;
    }
  }
}

// ---------------- small register-path WMMA GEMM (x_proj, N=48) ----------------
template<int NT>
__global__ void gemm_bf16_kernel(const __bf16* __restrict__ A, const __bf16* __restrict__ Bt,
                                 float* __restrict__ C, int M, int N, int K) {
  int wave = blockIdx.x * (blockDim.x >> 5) + (threadIdx.x >> 5);
  int nGroups = N / (16 * NT);
  int total = (M / 16) * nGroups;
  if (wave >= total) return;
  int mt   = wave / nGroups;
  int ng   = wave % nGroups;
  int lane = threadIdx.x & 31;
  int h    = lane >> 4;
  int nlo  = lane & 15;

  v8f acc[NT];
  #pragma unroll
  for (int t = 0; t < NT; t++)
    #pragma unroll
    for (int i = 0; i < 8; i++) acc[t][i] = 0.f;

  const __bf16* aRow = A + (size_t)(mt * 16 + nlo) * K;
  int colBase = ng * 16 * NT;

  for (int k0 = 0; k0 < K; k0 += 32) {
    v8bf alo = *(const v8bf*)(aRow + k0 + 8 * h);
    v8bf ahi = *(const v8bf*)(aRow + k0 + 16 + 8 * h);
    v16bf afrag;
    #pragma unroll
    for (int i = 0; i < 8; i++) { afrag[i] = alo[i]; afrag[8 + i] = ahi[i]; }
    #pragma unroll
    for (int t = 0; t < NT; t++) {
      v16bf bfrag;
      #pragma unroll
      for (int e = 0; e < 8; e++) {
        bfrag[e]     = Bt[(size_t)(k0 + 8 * h + e) * N + colBase + t * 16 + nlo];
        bfrag[8 + e] = Bt[(size_t)(k0 + 16 + 8 * h + e) * N + colBase + t * 16 + nlo];
      }
      acc[t] = __builtin_amdgcn_wmma_f32_16x16x32_bf16(
          false, afrag, false, bfrag, (short)0, acc[t], false, false);
    }
  }

  #pragma unroll
  for (int t = 0; t < NT; t++) {
    int n = colBase + t * 16 + nlo;
    #pragma unroll
    for (int r = 0; r < 8; r++) {
      int m = mt * 16 + 8 * h + r;
      C[(size_t)m * N + n] = acc[t][r];
    }
  }
}

// ---------------- causal depthwise conv + SiLU ----------------
__global__ void conv_silu_kernel(const float* __restrict__ xz, const float* __restrict__ conv_w,
                                 const float* __restrict__ conv_b,
                                 float* __restrict__ xconv, __bf16* __restrict__ xconvb) {
  int i = blockIdx.x * blockDim.x + threadIdx.x;
  if (i >= ROWS * DINNER) return;
  int d    = i % DINNER;
  int row  = i / DINNER;
  int t    = row % SEQ;
  int bidx = row / SEQ;
  float acc = conv_b[d];
  #pragma unroll
  for (int k = 0; k < DCONV; k++) {
    int tt = t - (DCONV - 1) + k;
    if (tt >= 0)
      acc += conv_w[d * DCONV + k] * xz[(size_t)(bidx * SEQ + tt) * (2 * DINNER) + d];
  }
  float s = acc / (1.f + __expf(-acc));
  xconv[i]  = s;
  xconvb[i] = (__bf16)s;
}

// ---------------- delta = softplus(dt @ W_dt^T + b_dt) ----------------
__global__ void delta_kernel(const float* __restrict__ xdbl, const float* __restrict__ Wdt,
                             const float* __restrict__ bdt, float* __restrict__ delta) {
  __shared__ float dts[DTRANK];
  int row   = blockIdx.x >> 1;
  int dbase = (blockIdx.x & 1) * 256;
  int tid   = threadIdx.x;
  if (tid < DTRANK) dts[tid] = xdbl[(size_t)row * XCOLS + tid];
  __syncthreads();
  int d = dbase + tid;
  float acc = bdt[d];
  #pragma unroll
  for (int r = 0; r < DTRANK; r++) acc += dts[r] * Wdt[d * DTRANK + r];
  float sp = (acc > 20.f) ? acc : log1pf(__expf(acc));
  delta[(size_t)row * DINNER + d] = sp;
}

// ---------------- sequential selective scan ----------------
#define TCHUNK 32
__global__ void scan_kernel(const float* __restrict__ xdbl, const float* __restrict__ delta,
                            const float* __restrict__ xconv, const float* __restrict__ xz,
                            const float* __restrict__ A_log, const float* __restrict__ Dskip,
                            __bf16* __restrict__ ybf) {
  __shared__ float BC[TCHUNK][2 * DSTATE];
  int b = blockIdx.x >> 1;
  int d = (blockIdx.x & 1) * 256 + threadIdx.x;
  float Ad[DSTATE], hn[DSTATE];
  #pragma unroll
  for (int n = 0; n < DSTATE; n++) { Ad[n] = -__expf(A_log[d * DSTATE + n]); hn[n] = 0.f; }
  float dsk = Dskip[d];

  for (int tc = 0; tc < SEQ / TCHUNK; tc++) {
    __syncthreads();
    for (int i = threadIdx.x; i < TCHUNK * 2 * DSTATE; i += 256) {
      int tt = i >> 5, c = i & 31;
      BC[tt][c] = xdbl[(size_t)(b * SEQ + tc * TCHUNK + tt) * XCOLS + DTRANK + c];
    }
    __syncthreads();
    for (int tt = 0; tt < TCHUNK; tt++) {
      size_t row = (size_t)b * SEQ + tc * TCHUNK + tt;
      float del = delta[row * DINNER + d];
      float xv  = xconv[row * DINNER + d];
      float zv  = xz[row * (2 * DINNER) + DINNER + d];
      float acc = 0.f;
      #pragma unroll
      for (int n = 0; n < DSTATE; n++) {
        float dA = __expf(del * Ad[n]);
        hn[n] = dA * hn[n] + (del * BC[tt][n]) * xv;
        acc  += hn[n] * BC[tt][DSTATE + n];
      }
      float y  = acc + xv * dsk;
      float sz = zv / (1.f + __expf(-zv));
      ybf[row * DINNER + d] = (__bf16)(y * sz);
    }
  }
}

// ---------------- BatchNorm over (B,L), per channel ----------------
__global__ void bn_kernel(const float* __restrict__ pre, const float* __restrict__ gamma,
                          const float* __restrict__ beta, float* __restrict__ out) {
  __shared__ float s1[256], s2[256];
  int c = blockIdx.x, tid = threadIdx.x;
  float a = 0.f, b = 0.f;
  for (int m = tid; m < ROWS; m += 256) {
    float v = pre[(size_t)m * DMODEL + c];
    a += v; b += v * v;
  }
  s1[tid] = a; s2[tid] = b; __syncthreads();
  for (int s = 128; s > 0; s >>= 1) {
    if (tid < s) { s1[tid] += s1[tid + s]; s2[tid] += s2[tid + s]; }
    __syncthreads();
  }
  float mean  = s1[0] / (float)ROWS;
  float var   = s2[0] / (float)ROWS - mean * mean;
  float scale = rsqrtf(var + EPSV) * gamma[c];
  float bet   = beta[c];
  for (int m = tid; m < ROWS; m += 256) {
    size_t idx = (size_t)m * DMODEL + c;
    out[idx] = (pre[idx] - mean) * scale + bet;
  }
}

// ---------------- launch ----------------
extern "C" void kernel_launch(void* const* d_in, const int* in_sizes, int n_in,
                              void* d_out, int out_size, void* d_ws, size_t ws_size,
                              hipStream_t stream) {
  const float* src    = (const float*)d_in[0];
  const float* W_in   = (const float*)d_in[1];
  const float* conv_w = (const float*)d_in[2];
  const float* conv_b = (const float*)d_in[3];
  const float* W_x    = (const float*)d_in[4];
  const float* W_dt   = (const float*)d_in[5];
  const float* b_dt   = (const float*)d_in[6];
  const float* A_log  = (const float*)d_in[7];
  const float* Dskip  = (const float*)d_in[8];
  const float* W_out  = (const float*)d_in[9];
  const float* gamma  = (const float*)d_in[10];
  const float* beta   = (const float*)d_in[11];
  float* out = (float*)d_out;

  char* p = (char*)d_ws;
  auto carve = [&](size_t bytes) -> char* {
    uintptr_t q = ((uintptr_t)p + 255) & ~(uintptr_t)255;
    p = (char*)(q + bytes);
    return (char*)q;
  };
  __bf16* srcb   = (__bf16*)carve((size_t)ROWS * DMODEL * 2);
  __bf16* Wint   = (__bf16*)carve((size_t)DMODEL * (2 * DINNER) * 2);
  __bf16* Wxt    = (__bf16*)carve((size_t)DINNER * XCOLS * 2);
  __bf16* Wot    = (__bf16*)carve((size_t)DINNER * DMODEL * 2);
  float*  xz     = (float*)carve((size_t)ROWS * (2 * DINNER) * 4);
  float*  xconv  = (float*)carve((size_t)ROWS * DINNER * 4);
  __bf16* xconvb = (__bf16*)carve((size_t)ROWS * DINNER * 2);
  float*  xdbl   = (float*)carve((size_t)ROWS * XCOLS * 4);
  float*  deltab = (float*)carve((size_t)ROWS * DINNER * 4);
  __bf16* ybf    = (__bf16*)carve((size_t)ROWS * DINNER * 2);
  float*  pre    = (float*)carve((size_t)ROWS * DMODEL * 4);

  cvt_bf16_kernel<<<(ROWS * DMODEL + 255) / 256, 256, 0, stream>>>(src, srcb, ROWS * DMODEL);
  cvt_tr_bf16_kernel<<<((2 * DINNER) * DMODEL + 255) / 256, 256, 0, stream>>>(W_in, Wint, 2 * DINNER, DMODEL);
  cvt_tr_bf16_kernel<<<(XCOLS * DINNER + 255) / 256, 256, 0, stream>>>(W_x, Wxt, XCOLS, DINNER);
  cvt_tr_bf16_kernel<<<(DMODEL * DINNER + 255) / 256, 256, 0, stream>>>(W_out, Wot, DMODEL, DINNER);

  // in_proj: [8192,256] x [256,1024]  -> 64 x 16 blocks of 128x64
  gemm_lds_kernel<DMODEL, false><<<(ROWS / 128) * ((2 * DINNER) / 64), 256, 0, stream>>>(
      srcb, Wint, xz, nullptr, ROWS, 2 * DINNER);

  conv_silu_kernel<<<(ROWS * DINNER + 255) / 256, 256, 0, stream>>>(xz, conv_w, conv_b, xconv, xconvb);

  // x_proj: [8192,512] x [512,48]
  gemm_bf16_kernel<3><<<(ROWS / 16) / 8, 256, 0, stream>>>(xconvb, Wxt, xdbl, ROWS, XCOLS, DINNER);

  delta_kernel<<<ROWS * 2, 256, 0, stream>>>(xdbl, W_dt, b_dt, deltab);
  scan_kernel<<<BATCH * 2, 256, 0, stream>>>(xdbl, deltab, xconv, xz, A_log, Dskip, ybf);

  // out_proj + residual: [8192,512] x [512,256] -> 64 x 4 blocks of 128x64
  gemm_lds_kernel<DINNER, true><<<(ROWS / 128) * (DMODEL / 64), 256, 0, stream>>>(
      ybf, Wot, pre, src, ROWS, DMODEL);

  bn_kernel<<<DMODEL, 256, 0, stream>>>(pre, gamma, beta, out);
}
